// MGU_20160576487855
// MI455X (gfx1250) — matrix operand order repaired and checked by reading, not compiled, not run on hardware
//
#include <hip/hip_runtime.h>

#define B_ 256
#define T_ 512
#define E_ 256
#define H_ 256

typedef __bf16 bf16_t;
typedef __attribute__((ext_vector_type(16))) __bf16 v16bf;
typedef __attribute__((ext_vector_type(8)))  __bf16 v8bf;
typedef __attribute__((ext_vector_type(8)))  float  v8f;
typedef __attribute__((ext_vector_type(4)))  float  v4f;

#define LDA 272   // padded bf16 row stride (stride%64dw==8 -> good bank pattern)
#define LDW 272
#define MT_ 4     // M-tiles (64 rows) per phase-1 block

// ---------- fragment loaders ----------

// A-matrix 16x32 bf16 fragment: lane holds row (lane&15);
// elems 0..15 = K[kb..kb+15], kb = kc*32 + (lane>>4)*16
__device__ __forceinline__ v16bf ld_frag_a_lds(const bf16_t* p) {
    v8bf lo = *(const v8bf*)p;
    v8bf hi = *(const v8bf*)(p + 8);
    v16bf r;
#pragma unroll
    for (int i = 0; i < 8; ++i) { r[i] = lo[i]; r[i + 8] = hi[i]; }
    return r;
}

// B-matrix 32x16 bf16 fragment: lane holds column (lane&15);
// 16 contiguous K values starting at kb = kc*32 + (lane>>4)*16
__device__ __forceinline__ v16bf ld_frag_b_lds(const bf16_t* p) {
    v8bf lo = *(const v8bf*)p;
    v8bf hi = *(const v8bf*)(p + 8);
    v16bf r;
#pragma unroll
    for (int i = 0; i < 8; ++i) { r[i] = lo[i]; r[i + 8] = hi[i]; }
    return r;
}

// Same B-fragment but sourced from row-major f32 global (converted inline)
__device__ __forceinline__ v16bf ld_frag_b_f32(const float* p) {
    v4f a = *(const v4f*)(p);
    v4f b = *(const v4f*)(p + 4);
    v4f c = *(const v4f*)(p + 8);
    v4f d = *(const v4f*)(p + 12);
    v16bf r;
#pragma unroll
    for (int i = 0; i < 4; ++i) {
        r[i]      = (bf16_t)a[i];
        r[i + 4]  = (bf16_t)b[i];
        r[i + 8]  = (bf16_t)c[i];
        r[i + 12] = (bf16_t)d[i];
    }
    return r;
}

__device__ __forceinline__ float fast_sigmoid(float x) {
    return 1.0f / (1.0f + __expf(-x));
}
__device__ __forceinline__ float fast_tanh(float x) {
    float e = __expf(2.0f * x);           // stable at +/-inf
    return 1.0f - 2.0f / (e + 1.0f);
}

// =====================================================================
// Phase 1: G[B*T, H] = sigmoid(X[B*T, E] @ Wx^T + bx), written into d_out
// One block = 64 rows of X (staged to LDS as bf16) x all 256 columns.
// Each wave owns one N-tile-pair across 4 M-tiles, so each converted Wx
// fragment is reused 4x (4x less L2 traffic + 4x fewer cvt ops).
// =====================================================================
__global__ void __launch_bounds__(256)
mgu_gates_kernel(const float* __restrict__ X, const float* __restrict__ Wx,
                 const float* __restrict__ Wxb, float* __restrict__ G)
{
    __shared__ __align__(16) bf16_t Alds[64 * LDA];   // ~34 KB

    const int tid   = threadIdx.x;
    const int wave  = tid >> 5;
    const int lane  = tid & 31;
    const int halfk = lane >> 4;
    const int l16   = lane & 15;
    const long mBase = (long)blockIdx.x * 64;

    // cooperative stage: 64 x 256 f32 -> bf16 (each thread: 64 contiguous elems)
    {
        const int r  = tid >> 2;          // 0..63
        const int cc = (tid & 3) << 6;    // 0,64,128,192
        const float* src = X + (mBase + r) * (long)E_ + cc;
        bf16_t* dst = Alds + r * LDA + cc;
#pragma unroll
        for (int q = 0; q < 16; ++q) {
            v4f x = *(const v4f*)(src + q * 4);
            dst[q * 4 + 0] = (bf16_t)x[0];
            dst[q * 4 + 1] = (bf16_t)x[1];
            dst[q * 4 + 2] = (bf16_t)x[2];
            dst[q * 4 + 3] = (bf16_t)x[3];
        }
    }
    __syncthreads();

    const int c0 = wave * 32 + l16;   // lane's two output columns
    const int c1 = c0 + 16;

    v8f acc0[MT_], acc1[MT_];
#pragma unroll
    for (int m = 0; m < MT_; ++m) {
        acc0[m] = (v8f){0.f,0.f,0.f,0.f,0.f,0.f,0.f,0.f};
        acc1[m] = (v8f){0.f,0.f,0.f,0.f,0.f,0.f,0.f,0.f};
    }

#pragma unroll
    for (int kc = 0; kc < 8; ++kc) {
        const int kbA = kc * 32 + halfk * 16;
        const int kbB = kc * 32 + halfk * 16;
        v16bf b0 = ld_frag_b_f32(Wx + (long)c0 * E_ + kbB);
        v16bf b1 = ld_frag_b_f32(Wx + (long)c1 * E_ + kbB);
#pragma unroll
        for (int m = 0; m < MT_; ++m) {
            v16bf a = ld_frag_a_lds(Alds + (m * 16 + l16) * LDA + kbA);
            acc0[m] = __builtin_amdgcn_wmma_f32_16x16x32_bf16(
                false, a, false, b0, (short)0, acc0[m], false, false);
            acc1[m] = __builtin_amdgcn_wmma_f32_16x16x32_bf16(
                false, a, false, b1, (short)0, acc1[m], false, false);
        }
    }

    const float bb0 = Wxb[c0];
    const float bb1 = Wxb[c1];
#pragma unroll
    for (int m = 0; m < MT_; ++m) {
        float* outp = G + (mBase + m * 16 + halfk * 8) * (long)H_;
#pragma unroll
        for (int v = 0; v < 8; ++v) {
            outp[(long)v * H_ + c0] = fast_sigmoid(acc0[m][v] + bb0);
            outp[(long)v * H_ + c1] = fast_sigmoid(acc1[m][v] + bb1);
        }
    }
}

// =====================================================================
// Phase 2: sequential scan. One block owns 16 batch rows for all T steps.
// Wh staged once in LDS as bf16 (~136KB, fits CDNA5's 320KB/WGP).
// h state lives in registers in WMMA C-layout; a double-buffered bf16
// shadow in LDS feeds the next step's A fragments -> ONE barrier/step.
// K-reduction split over two accumulator chains (even/odd kc) to halve
// the dependent-WMMA critical path. Gates are read from d_out and
// overwritten in place with the h outputs.
// =====================================================================
__global__ void __launch_bounds__(256)
mgu_scan_kernel(const float* __restrict__ Wh, const float* __restrict__ Whb,
                float* __restrict__ GO)
{
    __shared__ __align__(16) bf16_t WhL[H_ * LDW];    // 139264 B
    __shared__ __align__(16) bf16_t hL[2][16 * LDW];  // 2 x 8704 B

    const int tid   = threadIdx.x;
    const int wave  = tid >> 5;
    const int lane  = tid & 31;
    const int halfk = lane >> 4;
    const int l16   = lane & 15;
    const int bBase = blockIdx.x * 16;

    // stage Wh (row-major; column n of B == row n of Wh) as bf16
    {
        const float* src = Wh + (long)tid * H_;
        bf16_t* dst = WhL + tid * LDW;
#pragma unroll 4
        for (int c = 0; c < H_; c += 4) {
            v4f x = *(const v4f*)(src + c);
            dst[c + 0] = (bf16_t)x[0];
            dst[c + 1] = (bf16_t)x[1];
            dst[c + 2] = (bf16_t)x[2];
            dst[c + 3] = (bf16_t)x[3];
        }
    }
    for (int i = tid; i < 16 * LDW; i += 256) hL[0][i] = (bf16_t)0.0f;
    __syncthreads();

    const int c0 = wave * 32 + l16;   // this lane's two output columns
    const int c1 = c0 + 16;
    const float bb0 = Whb[c0];
    const float bb1 = Whb[c1];

    // per-lane global base: batch rows (bBase + halfk*8 + v), v = 0..7
    float* gbase = GO + ((long)(bBase + halfk * 8) * T_) * H_;
    const long rowStride = (long)T_ * H_;

    float h0[8], h1[8];
#pragma unroll
    for (int v = 0; v < 8; ++v) { h0[v] = 0.f; h1[v] = 0.f; }

    int cur = 0;
    for (int t = 0; t < T_; ++t) {
        const bf16_t* hRead = hL[cur];
        bf16_t* hWrite      = hL[cur ^ 1];

        // prefetch this step's gates (hidden behind the WMMA chain)
        float* gp = gbase + (long)t * H_;
        float g0[8], g1[8];
#pragma unroll
        for (int v = 0; v < 8; ++v) {
            g0[v] = gp[(long)v * rowStride + c0];
            g1[v] = gp[(long)v * rowStride + c1];
        }

        // two independent accumulator chains per tile (bias folded into a00/a10)
        v8f a00 = (v8f){bb0,bb0,bb0,bb0,bb0,bb0,bb0,bb0};
        v8f a10 = (v8f){bb1,bb1,bb1,bb1,bb1,bb1,bb1,bb1};
        v8f a01 = (v8f){0.f,0.f,0.f,0.f,0.f,0.f,0.f,0.f};
        v8f a11 = (v8f){0.f,0.f,0.f,0.f,0.f,0.f,0.f,0.f};
#pragma unroll
        for (int kc = 0; kc < 8; ++kc) {
            const int kbA = kc * 32 + halfk * 16;
            const int kbB = kc * 32 + halfk * 16;
            v16bf a  = ld_frag_a_lds(hRead + l16 * LDW + kbA);
            v16bf b0 = ld_frag_b_lds(WhL + c0 * LDW + kbB);
            v16bf b1 = ld_frag_b_lds(WhL + c1 * LDW + kbB);
            if (kc & 1) {
                a01 = __builtin_amdgcn_wmma_f32_16x16x32_bf16(
                    false, a, false, b0, (short)0, a01, false, false);
                a11 = __builtin_amdgcn_wmma_f32_16x16x32_bf16(
                    false, a, false, b1, (short)0, a11, false, false);
            } else {
                a00 = __builtin_amdgcn_wmma_f32_16x16x32_bf16(
                    false, a, false, b0, (short)0, a00, false, false);
                a10 = __builtin_amdgcn_wmma_f32_16x16x32_bf16(
                    false, a, false, b1, (short)0, a10, false, false);
            }
        }
        v8f s0 = a00 + a01;
        v8f s1 = a10 + a11;

#pragma unroll
        for (int v = 0; v < 8; ++v) {
            float p0  = fast_tanh(s0[v]);
            float p1  = fast_tanh(s1[v]);
            float hn0 = h0[v] * g0[v] + (1.f - g0[v]) * p0;
            float hn1 = h1[v] * g1[v] + (1.f - g1[v]) * p1;
            gp[(long)v * rowStride + c0] = hn0;   // overwrite gate with h_t
            gp[(long)v * rowStride + c1] = hn1;
            h0[v] = hn0;
            h1[v] = hn1;
            const int r = v + halfk * 8;
            hWrite[r * LDW + c0] = (bf16_t)hn0;
            hWrite[r * LDW + c1] = (bf16_t)hn1;
        }
        __syncthreads();   // hWrite complete & visible before next step reads it
        cur ^= 1;
    }
}

extern "C" void kernel_launch(void* const* d_in, const int* in_sizes, int n_in,
                              void* d_out, int out_size, void* d_ws, size_t ws_size,
                              hipStream_t stream) {
    (void)in_sizes; (void)n_in; (void)out_size; (void)d_ws; (void)ws_size;
    const float* x   = (const float*)d_in[0];
    const float* Wxw = (const float*)d_in[1];
    const float* Wxb = (const float*)d_in[2];
    const float* Whw = (const float*)d_in[3];
    const float* Whb = (const float*)d_in[4];
    float* out = (float*)d_out;

    mgu_gates_kernel<<<dim3((B_ * T_) / 64), dim3(256), 0, stream>>>(x, Wxw, Wxb, out);
    mgu_scan_kernel<<<dim3(B_ / 16), dim3(256), 0, stream>>>(Whw, Whb, out);
}
